// QEMFormer_18708877541594
// MI455X (gfx1250) — matrix operand order repaired and checked by reading, not compiled, not run on hardware
//
#include <hip/hip_runtime.h>
#include <hip/hip_bf16.h>
#include <math.h>

#define N_NODES 50000
#define N_PAD   50048          // 391 * 128 : padded so every WMMA tile is full
#define N_EDGES 1200000
#define N_GRAPHS 256
#define NODE_DIM 64
#define HEADS 4
#define HEAD_DIM 16
#define CTX_DIM 3

typedef __attribute__((ext_vector_type(16))) _Float16 v16h;
typedef __attribute__((ext_vector_type(4)))  _Float16 v4h;
typedef __attribute__((ext_vector_type(8)))  float    v8f;

union CF { v8f v; float e[8]; };

// ---------- float <-> ordered-uint encoding for atomic max on floats ----------
__device__ __forceinline__ unsigned encOrd(float f) {
  unsigned b = __float_as_uint(f);
  return (b & 0x80000000u) ? ~b : (b | 0x80000000u);
}
__device__ __forceinline__ float decOrd(unsigned u) {
  unsigned b = (u & 0x80000000u) ? (u & 0x7fffffffu) : ~u;
  return __uint_as_float(b);
}

// ---------------------------------- utility ----------------------------------
__global__ void k_fill0(float* p, long long n) {
  long long i = (long long)blockIdx.x * blockDim.x + threadIdx.x;
  if (i < n) p[i] = 0.0f;
}

// ------------------------- h = emb[gid] + p*Wp + bp ---------------------------
__global__ void k_init_h(const float* __restrict__ x, const float* __restrict__ emb,
                         const float* __restrict__ Wp, const float* __restrict__ bp,
                         float* __restrict__ h) {
  int t = blockIdx.x * blockDim.x + threadIdx.x;
  if (t >= N_NODES * NODE_DIM) return;
  int n = t >> 6, d = t & 63;
  int g = (int)x[n * 2 + 0];
  float p = x[n * 2 + 1];
  h[t] = emb[g * NODE_DIM + d] + p * Wp[d] + bp[d];
}

// ------------- WMMA: q,k,v,skip = h @ W* + b* (per wave: 16 nodes) ------------
//
// All node buffers are padded to N_PAD = 391*128, so every tile is full:
// no bounds checks in staging or writeback.
//
// LDS holds fragment-ordered data so every lane's A/B fragment is 16 contiguous
// f16 (=> two ds_load_b128 per fragment):
//   A (16x32 f16): lane half 0 holds K 0..15, half 1 holds K 16..31
//   B (32x16 f16): lane half 0 holds K 0..15, half 1 holds K 16..31
//   D (16x16 f32): column = lane&15, row = vgpr + 8*(lane>>4)
__global__ void __launch_bounds__(256)
k_qkvs_wmma(const float* __restrict__ h,
            const float* __restrict__ Wq, const float* __restrict__ Wk,
            const float* __restrict__ Wv, const float* __restrict__ Wsk,
            const float* __restrict__ bq, const float* __restrict__ bk,
            const float* __restrict__ bv, const float* __restrict__ bsk,
            float* __restrict__ oq, float* __restrict__ ok,
            float* __restrict__ ov, float* __restrict__ osk) {
  // sH[row][group(2)][half(2)][idx(16)] : fragment-ordered h tile (16 KB)
  __shared__ __attribute__((aligned(32))) _Float16 sH[128 * 64];
  // sW[w(4)][ks(2)][col(64)][kk(32)]    : fragment-ordered (transposed) W (32 KB)
  __shared__ __attribute__((aligned(32))) _Float16 sW[4 * 2 * 64 * 32];

  int tid  = threadIdx.x;
  int base = blockIdx.x * 128;

  // ---- stage h tile: float4 global loads -> swizzled v4h LDS stores ----
  for (int i = tid; i < 128 * 16; i += 256) {
    int r  = i >> 4;
    int c4 = (i & 15) * 4;                  // starting channel (multiple of 4)
    float4 val = *(const float4*)(h + (base + r) * NODE_DIM + c4);
    int grp = c4 >> 5;                      // K-step (0: K<32, 1: K>=32)
    int kk  = c4 & 31;
    int hf  = kk >> 4;                      // lane half owning this K
    int idx = kk & 15;
    v4h pk = { (_Float16)val.x, (_Float16)val.y, (_Float16)val.z, (_Float16)val.w };
    *(v4h*)&sH[(r * 2 + grp) * 32 + hf * 16 + idx] = pk;
  }

  // ---- stage weights transposed: sW[((w*2+ks)*64+n)*32 + kk] = W[ks*32+kk][n]
  const float* Wsrc[4] = {Wq, Wk, Wv, Wsk};
  #pragma unroll
  for (int w = 0; w < 4; ++w) {
    for (int i = tid; i < 1024; i += 256) { // (k-quad, n)
      int n  = i & 63;
      int k4 = (i >> 6) * 4;
      const float* src = Wsrc[w] + k4 * NODE_DIM + n;   // coalesced in n
      v4h pk = { (_Float16)src[0], (_Float16)src[64],
                 (_Float16)src[128], (_Float16)src[192] };
      int ks = k4 >> 5;
      *(v4h*)&sW[(((w * 2 + ks) * 64) + n) * 32 + (k4 & 31)] = pk;
    }
  }
  __syncthreads();

  int wv = tid >> 5, lane = tid & 31;
  int half = lane >> 4, lr = lane & 15;
  int mrow = wv * 16 + lr;

  v16h a0 = *(const v16h*)&sH[(mrow * 2 + 0) * 32 + half * 16];
  v16h a1 = *(const v16h*)&sH[(mrow * 2 + 1) * 32 + half * 16];

  const float* Bias[4] = {bq, bk, bv, bsk};
  float*       Outp[4] = {oq, ok, ov, osk};
  int nodeBase = base + wv * 16 + 8 * half;

  #pragma unroll
  for (int w = 0; w < 4; ++w) {
    #pragma unroll
    for (int nt = 0; nt < 4; ++nt) {
      int col = nt * 16 + lr;
      v16h b0 = *(const v16h*)&sW[((w * 2 + 0) * 64 + col) * 32 + half * 16];
      v16h b1 = *(const v16h*)&sW[((w * 2 + 1) * 64 + col) * 32 + half * 16];
      // two independent accumulators: breaks the WMMA->WMMA RAW chain so the
      // scheduler can interleave tiles instead of inserting hazard NOPs
      CF acc0; acc0.v = (v8f){0.f,0.f,0.f,0.f,0.f,0.f,0.f,0.f};
      CF acc1; acc1.v = (v8f){0.f,0.f,0.f,0.f,0.f,0.f,0.f,0.f};
      acc0.v = __builtin_amdgcn_wmma_f32_16x16x32_f16(false, a0, false, b0,
                                                      (short)0, acc0.v, false, false);
      acc1.v = __builtin_amdgcn_wmma_f32_16x16x32_f16(false, a1, false, b1,
                                                      (short)0, acc1.v, false, false);
      float bias = Bias[w][col];
      int idx0 = nodeBase * NODE_DIM + col;          // 32-bit, saddr-friendly
      #pragma unroll
      for (int r = 0; r < 8; ++r)
        Outp[w][idx0 + r * NODE_DIM] = acc0.e[r] + acc1.e[r] + bias;
    }
  }
}

// ----------------------- edge pass 1: score + segment max ---------------------
__global__ void k_edge_score(const float* __restrict__ q, const float* __restrict__ kk,
                             const int* __restrict__ ei, float* __restrict__ scores,
                             unsigned* __restrict__ mEnc) {
  long long t = (long long)blockIdx.x * blockDim.x + threadIdx.x;
  if (t >= (long long)N_EDGES * HEADS) return;
  int e = (int)(t >> 2), hd = (int)(t & 3);
  int s = ei[e], d = ei[N_EDGES + e];
  const float4* qp = (const float4*)(q  + d * NODE_DIM + hd * HEAD_DIM);
  const float4* kp = (const float4*)(kk + s * NODE_DIM + hd * HEAD_DIM);
  float sc = 0.f;
  #pragma unroll
  for (int i = 0; i < 4; ++i) {
    float4 a = qp[i], b = kp[i];
    sc += a.x * b.x + a.y * b.y + a.z * b.z + a.w * b.w;
  }
  sc *= 0.25f;                                   // 1/sqrt(HEAD_DIM)
  scores[e * HEADS + hd] = sc;
  atomicMax(&mEnc[d * HEADS + hd], encOrd(sc));
}

// ---------------------- edge pass 2: exp + denominator ------------------------
__global__ void k_edge_exp(const int* __restrict__ ei, const unsigned* __restrict__ mEnc,
                           float* __restrict__ scores, float* __restrict__ denom) {
  long long t = (long long)blockIdx.x * blockDim.x + threadIdx.x;
  if (t >= (long long)N_EDGES * HEADS) return;
  int e = (int)(t >> 2), hd = (int)(t & 3);
  int d = ei[N_EDGES + e];
  float m  = decOrd(mEnc[d * HEADS + hd]);
  float ex = __expf(scores[e * HEADS + hd] - m);
  scores[e * HEADS + hd] = ex;
  atomicAdd(&denom[d * HEADS + hd], ex);
}

// ------- edge pass 3: scatter alpha * v[src] into agg (thread = edge,head) ----
__global__ void k_edge_agg(const int* __restrict__ ei, const float* __restrict__ scores,
                           const float* __restrict__ denom, const float* __restrict__ v,
                           float* __restrict__ agg) {
  long long t = (long long)blockIdx.x * blockDim.x + threadIdx.x;
  if (t >= (long long)N_EDGES * HEADS) return;
  int e = (int)(t >> 2), hd = (int)(t & 3);
  int s = ei[e], d = ei[N_EDGES + e];
  float alpha = scores[e * HEADS + hd] / denom[d * HEADS + hd];
  const float4* vp = (const float4*)(v + s * NODE_DIM + hd * HEAD_DIM);
  float* ap = agg + d * NODE_DIM + hd * HEAD_DIM;
  #pragma unroll
  for (int i = 0; i < 4; ++i) {
    float4 vv = vp[i];
    atomicAdd(&ap[i * 4 + 0], alpha * vv.x);
    atomicAdd(&ap[i * 4 + 1], alpha * vv.y);
    atomicAdd(&ap[i * 4 + 2], alpha * vv.z);
    atomicAdd(&ap[i * 4 + 3], alpha * vv.w);
  }
}

// ------------------------- h = relu(agg + skip) (float4) ----------------------
__global__ void k_combine_relu(const float* __restrict__ agg, const float* __restrict__ sk,
                               float* __restrict__ h) {
  int t = blockIdx.x * blockDim.x + threadIdx.x;
  if (t >= N_NODES * NODE_DIM / 4) return;
  float4 a = ((const float4*)agg)[t];
  float4 b = ((const float4*)sk)[t];
  float4 r;
  r.x = fmaxf(a.x + b.x, 0.f); r.y = fmaxf(a.y + b.y, 0.f);
  r.z = fmaxf(a.z + b.z, 0.f); r.w = fmaxf(a.w + b.w, 0.f);
  ((float4*)h)[t] = r;
}

// ------------------------------- pooling --------------------------------------
__global__ void k_pool_sum(const float* __restrict__ h, const int* __restrict__ batch,
                           float* __restrict__ sums) {
  int t = blockIdx.x * blockDim.x + threadIdx.x;
  if (t >= N_NODES * NODE_DIM) return;
  int n = t >> 6, d = t & 63;
  atomicAdd(&sums[batch[n] * NODE_DIM + d], h[t]);
}
__global__ void k_pool_cnt(const int* __restrict__ batch, float* __restrict__ counts) {
  int n = blockIdx.x * blockDim.x + threadIdx.x;
  if (n >= N_NODES) return;
  atomicAdd(&counts[batch[n]], 1.0f);
}
__global__ void k_pool_div(const float* __restrict__ sums, const float* __restrict__ counts,
                           float* __restrict__ ge) {
  int t = blockIdx.x * blockDim.x + threadIdx.x;
  if (t >= N_GRAPHS * NODE_DIM) return;
  int g = t >> 6;
  ge[t] = sums[t] / fmaxf(counts[g], 1.0f);
}

// --------------------------- context + MLP head -------------------------------
__global__ void k_ctx(const float* __restrict__ ga, const float* __restrict__ Wc,
                      const float* __restrict__ bc, float* __restrict__ ctx) {
  int t = blockIdx.x * blockDim.x + threadIdx.x;
  if (t >= N_GRAPHS * NODE_DIM) return;
  int g = t >> 6, d = t & 63;
  float acc = bc[d];
  #pragma unroll
  for (int c = 0; c < CTX_DIM; ++c) acc += ga[g * CTX_DIM + c] * Wc[c * NODE_DIM + d];
  ctx[t] = fmaxf(acc, 0.f);
}
__global__ void k_mlp1(const float* __restrict__ ge, const float* __restrict__ ctx,
                       const float* __restrict__ W1, const float* __restrict__ b1,
                       float* __restrict__ o1) {
  int t = blockIdx.x * blockDim.x + threadIdx.x;
  if (t >= N_GRAPHS * 64) return;
  int g = t >> 6, j = t & 63;
  float acc = b1[j];
  for (int i = 0; i < 64; ++i) acc += ge[g * 64 + i]  * W1[i * 64 + j];
  for (int i = 0; i < 64; ++i) acc += ctx[g * 64 + i] * W1[(64 + i) * 64 + j];
  o1[t] = fmaxf(acc, 0.f);
}
__global__ void k_mlp2(const float* __restrict__ o1, const float* __restrict__ W2,
                       const float* __restrict__ b2, float* __restrict__ o2) {
  int t = blockIdx.x * blockDim.x + threadIdx.x;
  if (t >= N_GRAPHS * 32) return;
  int g = t >> 5, j = t & 31;
  float acc = b2[j];
  for (int i = 0; i < 64; ++i) acc += o1[g * 64 + i] * W2[i * 32 + j];
  o2[t] = fmaxf(acc, 0.f);
}
__global__ void k_mlp3(const float* __restrict__ o2, const float* __restrict__ W3,
                       const float* __restrict__ b3, float* __restrict__ out) {
  int g = blockIdx.x * blockDim.x + threadIdx.x;
  if (g >= N_GRAPHS) return;
  float acc = b3[0];
  for (int i = 0; i < 32; ++i) acc += o2[g * 32 + i] * W3[i];
  out[g] = acc;
}

// --------------------------------- launcher -----------------------------------
extern "C" void kernel_launch(void* const* d_in, const int* in_sizes, int n_in,
                              void* d_out, int out_size, void* d_ws, size_t ws_size,
                              hipStream_t stream) {
  const float* x     = (const float*)d_in[0];
  const int*   ei    = (const int*)  d_in[1];
  const int*   batch = (const int*)  d_in[2];
  const float* ga    = (const float*)d_in[3];
  const float* emb   = (const float*)d_in[4];
  const float* Wp    = (const float*)d_in[5];
  const float* bp    = (const float*)d_in[6];
  const float* Wq    = (const float*)d_in[7];
  const float* bq    = (const float*)d_in[8];
  const float* Wk    = (const float*)d_in[9];
  const float* bk    = (const float*)d_in[10];
  const float* Wv    = (const float*)d_in[11];
  const float* bv    = (const float*)d_in[12];
  const float* Wsk   = (const float*)d_in[13];
  const float* bsk   = (const float*)d_in[14];
  const float* Wc    = (const float*)d_in[15];
  const float* bc    = (const float*)d_in[16];
  const float* W1    = (const float*)d_in[17];
  const float* b1    = (const float*)d_in[18];
  const float* W2    = (const float*)d_in[19];
  const float* b2    = (const float*)d_in[20];
  const float* W3    = (const float*)d_in[21];
  const float* b3    = (const float*)d_in[22];
  float* out = (float*)d_out;

  // workspace carve-out (node buffers padded to N_PAD rows)
  char* ws = (char*)d_ws;
  size_t off = 0;
  auto carve = [&](size_t bytes) { char* p = ws + off; off += (bytes + 255) & ~(size_t)255; return p; };
  float*    h     = (float*)   carve((size_t)N_PAD * NODE_DIM * 4);
  float*    qbuf  = (float*)   carve((size_t)N_PAD * NODE_DIM * 4);
  float*    kbuf  = (float*)   carve((size_t)N_PAD * NODE_DIM * 4);
  float*    vbuf  = (float*)   carve((size_t)N_PAD * NODE_DIM * 4);
  float*    skbuf = (float*)   carve((size_t)N_PAD * NODE_DIM * 4);
  float*    agg   = (float*)   carve((size_t)N_PAD * NODE_DIM * 4);
  float*    sc    = (float*)   carve((size_t)N_EDGES * HEADS * 4);
  unsigned* mEnc  = (unsigned*)carve((size_t)N_NODES * HEADS * 4);
  float*    denom = (float*)   carve((size_t)N_NODES * HEADS * 4);
  float*    sums  = (float*)   carve((size_t)N_GRAPHS * NODE_DIM * 4);
  float*    cnts  = (float*)   carve((size_t)N_GRAPHS * 4);
  float*    ge    = (float*)   carve((size_t)N_GRAPHS * NODE_DIM * 4);
  float*    ctx   = (float*)   carve((size_t)N_GRAPHS * NODE_DIM * 4);
  float*    o1    = (float*)   carve((size_t)N_GRAPHS * 64 * 4);
  float*    o2    = (float*)   carve((size_t)N_GRAPHS * 32 * 4);

  #define GRID1(n) ((unsigned)(((long long)(n) + 255) / 256))

  // zero the padding rows of h once so no NaNs enter the padded WMMA tiles
  k_fill0<<<GRID1((long long)(N_PAD - N_NODES) * NODE_DIM), 256, 0, stream>>>(
      h + (size_t)N_NODES * NODE_DIM, (long long)(N_PAD - N_NODES) * NODE_DIM);
  k_init_h<<<GRID1((long long)N_NODES * NODE_DIM), 256, 0, stream>>>(x, emb, Wp, bp, h);

  for (int l = 0; l < 2; ++l) {
    const float* Wq_l  = Wq  + (size_t)l * NODE_DIM * NODE_DIM;
    const float* Wk_l  = Wk  + (size_t)l * NODE_DIM * NODE_DIM;
    const float* Wv_l  = Wv  + (size_t)l * NODE_DIM * NODE_DIM;
    const float* Wsk_l = Wsk + (size_t)l * NODE_DIM * NODE_DIM;
    const float* bq_l  = bq  + (size_t)l * NODE_DIM;
    const float* bk_l  = bk  + (size_t)l * NODE_DIM;
    const float* bv_l  = bv  + (size_t)l * NODE_DIM;
    const float* bsk_l = bsk + (size_t)l * NODE_DIM;

    k_qkvs_wmma<<<N_PAD / 128, 256, 0, stream>>>(
        h, Wq_l, Wk_l, Wv_l, Wsk_l, bq_l, bk_l, bv_l, bsk_l, qbuf, kbuf, vbuf, skbuf);

    k_fill0<<<GRID1((long long)N_NODES * HEADS), 256, 0, stream>>>((float*)mEnc, (long long)N_NODES * HEADS);
    k_fill0<<<GRID1((long long)N_NODES * HEADS), 256, 0, stream>>>(denom, (long long)N_NODES * HEADS);
    k_fill0<<<GRID1((long long)N_NODES * NODE_DIM), 256, 0, stream>>>(agg, (long long)N_NODES * NODE_DIM);

    k_edge_score<<<GRID1((long long)N_EDGES * HEADS), 256, 0, stream>>>(qbuf, kbuf, ei, sc, mEnc);
    k_edge_exp  <<<GRID1((long long)N_EDGES * HEADS), 256, 0, stream>>>(ei, mEnc, sc, denom);
    k_edge_agg  <<<GRID1((long long)N_EDGES * HEADS), 256, 0, stream>>>(ei, sc, denom, vbuf, agg);

    k_combine_relu<<<GRID1((long long)N_NODES * NODE_DIM / 4), 256, 0, stream>>>(agg, skbuf, h);
  }

  k_fill0<<<GRID1((long long)N_GRAPHS * NODE_DIM), 256, 0, stream>>>(sums, (long long)N_GRAPHS * NODE_DIM);
  k_fill0<<<GRID1((long long)N_GRAPHS), 256, 0, stream>>>(cnts, (long long)N_GRAPHS);

  k_pool_sum<<<GRID1((long long)N_NODES * NODE_DIM), 256, 0, stream>>>(h, batch, sums);
  k_pool_cnt<<<GRID1((long long)N_NODES), 256, 0, stream>>>(batch, cnts);
  k_pool_div<<<GRID1((long long)N_GRAPHS * NODE_DIM), 256, 0, stream>>>(sums, cnts, ge);

  k_ctx <<<GRID1((long long)N_GRAPHS * NODE_DIM), 256, 0, stream>>>(ga, Wc, bc, ctx);
  k_mlp1<<<GRID1((long long)N_GRAPHS * 64), 256, 0, stream>>>(ge, ctx, W1, b1, o1);
  k_mlp2<<<GRID1((long long)N_GRAPHS * 32), 256, 0, stream>>>(o1, W2, b2, o2);
  k_mlp3<<<GRID1((long long)N_GRAPHS), 256, 0, stream>>>(o2, W3, b3, out);

  #undef GRID1
}
